// RelativeAttention_16484084482887
// MI455X (gfx1250) — matrix operand order repaired and checked by reading, not compiled, not run on hardware
//
#include <hip/hip_runtime.h>

// XLNet/Transformer-XL relative attention for MI455X (gfx1250), wave32 + WMMA bf16.
// QLEN=KLEN=512, RLEN=1024, BSZ=8, NHEAD=16, DHEAD=64, SCALE=0.125.
// Flash-attention structure; BD term computed via sliding R-window WMMA tiles
// + LDS diagonal gather (rel_shift => BD[i,j] = Qr_i . R[512+j-i]).

#define QLEN 512
#define KLEN 512
#define RLEN 1024
#define BSZ 8
#define NHEAD 16
#define DHEAD 64
#define ROWSTR 8192            // BSZ*NHEAD*DHEAD
#define SCALEF 0.125f
#define BM 128                 // query rows per block (8 waves x 16)
#define BN 32                  // kv columns per step

typedef __attribute__((ext_vector_type(16))) __bf16 v16bf;
typedef __attribute__((ext_vector_type(8)))  float  v8f;

union Frag { unsigned u[8]; uint4 q2[2]; v16bf v; };

// f32 -> packed bf16 pair via CDNA5 V_FMA_MIXLO/MIXHI_BF16 (VOP3P, RNE convert).
__device__ __forceinline__ unsigned pkbf(float a, float b) {
  unsigned r;
  asm("v_fma_mixlo_bf16 %0, %1, 1.0, 0\n\t"
      "v_fma_mixhi_bf16 %0, %2, 1.0, 0"
      : "=&v"(r) : "v"(a), "v"(b));
  return r;
}
__device__ __forceinline__ unsigned short bf16b(float a) {
  unsigned r;
  asm("v_fma_mixlo_bf16 %0, %1, 1.0, 0" : "=v"(r) : "v"(a));
  return (unsigned short)r;
}

__device__ __forceinline__ v8f wmma_bf16(const Frag& a, const Frag& b, v8f c) {
  return __builtin_amdgcn_wmma_f32_16x16x32_bf16(false, a.v, false, b.v,
                                                 (short)0, c, false, false);
}

// B-operand fragment (32x16 bf16) from row-major packed LDS, row stride 36 dwords.
// Lane: N = lane%16; dwords hold K pairs: lane<16 -> K 0..15, lane>=16 -> K 16..31 (+32*h).
__device__ __forceinline__ Frag ldB(const unsigned* base, int row, int h, int lane) {
  const unsigned* p = base + row * 36 + 16 * h + ((lane & 16) ? 8 : 0);
  Frag f;
  f.q2[0] = *(const uint4*)p;
  f.q2[1] = *(const uint4*)(p + 4);
  return f;
}

// B-operand fragment for P@V from transposed V (VT[d][slot] bf16, dword stride 20).
__device__ __forceinline__ Frag ldV(const unsigned* vt, int t, int lane) {
  int d = 16 * t + (lane & 15);
  const unsigned* p = vt + d * 20 + ((lane & 16) ? 4 : 0);
  Frag f;
  f.q2[0] = *(const uint4*)p;
  f.q2[1] = *(const uint4*)(p + 8);
  return f;
}

// A-operand fragment (16x32 bf16) for P from row-major LDS (dword stride 20).
__device__ __forceinline__ Frag ldP(const unsigned* pt, int lane) {
  int r = lane & 15;
  const unsigned* p = pt + r * 20 + ((lane & 16) ? 4 : 0);
  Frag f;
  f.q2[0] = *(const uint4*)p;       // K slots 0..7   (or 8..15)
  f.q2[1] = *(const uint4*)(p + 8); // K slots 16..23 (or 24..31)
  return f;
}

__global__ __launch_bounds__(256) void xlnet_rel_attn(
    const float* __restrict__ q, const float* __restrict__ kh,
    const float* __restrict__ vh, const float* __restrict__ kr,
    const float* __restrict__ seg_embed, const unsigned char* __restrict__ seg_mat,
    const float* __restrict__ rwb, const float* __restrict__ rrb,
    const float* __restrict__ rsb, float* __restrict__ out)
{
  __shared__ __attribute__((aligned(16))) unsigned       ringS[256 * 36]; // R ring, bf16 pairs
  __shared__ __attribute__((aligned(16))) unsigned       Ksm[32 * 36];    // K tile, bf16 pairs
  __shared__ __attribute__((aligned(16))) unsigned short Vsm[64 * 40];    // V transposed bf16
  __shared__ __attribute__((aligned(16))) float          Wg[8][16 * 48];  // per-wave W gather
  __shared__ __attribute__((aligned(16))) unsigned       Pt[8][16 * 20];  // per-wave P staging
  __shared__ __attribute__((aligned(16))) float          efs[BM][2];

  const int I0 = blockIdx.x * BM;
  const int b = blockIdx.y, n = blockIdx.z;
  const int tid = threadIdx.x;
  const int wv = tid >> 5, lane = tid & 31;
  const int i0 = I0 + wv * 16;
  const int bn = (b * NHEAD + n) * DHEAD;

  // ---- EF precompute: one dot product per thread (256 = 128 rows x 2 segs)
  {
    int row = tid >> 1, s = tid & 1;
    const float* qp = q + (size_t)(I0 + row) * ROWSTR + bn;
    const float* sp = seg_embed + (s * NHEAD + n) * DHEAD;
    const float* bp = rsb + n * DHEAD;
    float acc = 0.f;
    #pragma unroll
    for (int d = 0; d < DHEAD; d += 4) {
      float4 qv = *(const float4*)(qp + d);
      float4 sv = *(const float4*)(sp + d);
      float4 bv = *(const float4*)(bp + d);
      acc += (qv.x + bv.x) * sv.x + (qv.y + bv.y) * sv.y +
             (qv.z + bv.z) * sv.z + (qv.w + bv.w) * sv.w;
    }
    efs[row][s] = acc;
  }

  // ---- Q A-fragments: (q + r_w_bias) and (q + r_r_bias), two K-halves each
  Frag aw[2], ar[2];
  {
    int r = lane & 15;
    const float* qp = q + (size_t)(i0 + r) * ROWSTR + bn;
    const float* wb = rwb + n * DHEAD;
    const float* rb = rrb + n * DHEAD;
    int hb = (lane & 16) ? 8 : 0;
    #pragma unroll
    for (int h = 0; h < 2; ++h) {
      #pragma unroll
      for (int v = 0; v < 8; ++v) {
        int d0 = 32 * h + 2 * hb + 2 * v;
        float2 qv = *(const float2*)(qp + d0);
        float2 w2 = *(const float2*)(wb + d0);
        float2 r2 = *(const float2*)(rb + d0);
        aw[h].u[v] = pkbf(qv.x + w2.x, qv.y + w2.y);
        ar[h].u[v] = pkbf(qv.x + r2.x, qv.y + r2.y);
      }
    }
  }

  // ---- Prologue: stage 128 R rows [385 - I0, 513 - I0) into the ring
  {
    int trow = tid >> 3, dd = (tid & 7) * 8;
    #pragma unroll
    for (int c = 0; c < 4; ++c) {
      int row = 385 - I0 + 32 * c + trow;
      const float* sp = kr + (size_t)row * ROWSTR + bn + dd;
      float4 f0 = *(const float4*)sp, f1 = *(const float4*)(sp + 4);
      unsigned* dp = ringS + (row & 255) * 36 + (dd >> 1);
      dp[0] = pkbf(f0.x, f0.y); dp[1] = pkbf(f0.z, f0.w);
      dp[2] = pkbf(f1.x, f1.y); dp[3] = pkbf(f1.z, f1.w);
    }
  }
  __syncthreads();

  // ---- Prologue W2: R rows [497 - i0, 497 - i0 + 16)
  v8f w2acc = {};
  {
    int c = lane & 15;
    int slot0 = (497 - i0 + c) & 255;
    Frag b0 = ldB(ringS, slot0, 0, lane);
    Frag b1 = ldB(ringS, slot0, 1, lane);
    w2acc = wmma_bf16(ar[0], b0, w2acc);
    w2acc = wmma_bf16(ar[1], b1, w2acc);
  }

  float mrow[8], lrow[8];
  #pragma unroll
  for (int v = 0; v < 8; ++v) { mrow[v] = -3.0e38f; lrow[v] = 0.f; }
  v8f o0 = {}, o1 = {}, o2 = {}, o3 = {};

  const int nsteps = I0 / 32 + 4;
  const unsigned* vtu = (const unsigned*)Vsm;
  const unsigned* ptu = (const unsigned*)Pt[wv];

  for (int step = 0; step < nsteps; ++step) {
    const int j0 = step * BN;

    // --- cooperative staging: K tile, transposed V tile, 32 new R ring rows
    {
      int trow = tid >> 3, dd = (tid & 7) * 8;
      { // K rows [j0, j0+32), row-major bf16
        const float* sp = kh + (size_t)(j0 + trow) * ROWSTR + bn + dd;
        float4 f0 = *(const float4*)sp, f1 = *(const float4*)(sp + 4);
        unsigned* dp = Ksm + trow * 36 + (dd >> 1);
        dp[0] = pkbf(f0.x, f0.y); dp[1] = pkbf(f0.z, f0.w);
        dp[2] = pkbf(f1.x, f1.y); dp[3] = pkbf(f1.z, f1.w);
      }
      { // V rows [j0, j0+32), transposed store VT[d][slot],
        // K-slot interleave: local col c -> slot 2c, col c+16 -> slot 2c+1
        const float* sp = vh + (size_t)(j0 + trow) * ROWSTR + bn + dd;
        float4 f0 = *(const float4*)sp, f1 = *(const float4*)(sp + 4);
        float fv[8] = {f0.x, f0.y, f0.z, f0.w, f1.x, f1.y, f1.z, f1.w};
        int slot = 2 * (trow & 15) + (trow >> 4);
        #pragma unroll
        for (int k2 = 0; k2 < 8; ++k2)
          Vsm[(dd + k2) * 40 + slot] = bf16b(fv[k2]);
      }
      { // R rows [513 + j0 - I0, +32)
        int row = 513 + j0 - I0 + trow;
        const float* sp = kr + (size_t)row * ROWSTR + bn + dd;
        float4 f0 = *(const float4*)sp, f1 = *(const float4*)(sp + 4);
        unsigned* dp = ringS + (row & 255) * 36 + (dd >> 1);
        dp[0] = pkbf(f0.x, f0.y); dp[1] = pkbf(f0.z, f0.w);
        dp[2] = pkbf(f1.x, f1.y); dp[3] = pkbf(f1.z, f1.w);
      }
      // --- prefetch next step's tiles into cache while this step computes
      if (step + 1 < nsteps) {
        int j0n = j0 + BN;
        __builtin_prefetch(kh + (size_t)(j0n + trow) * ROWSTR + bn + dd, 0, 1);
        __builtin_prefetch(vh + (size_t)(j0n + trow) * ROWSTR + bn + dd, 0, 1);
        __builtin_prefetch(kr + (size_t)(513 + j0n - I0 + trow) * ROWSTR + bn + dd, 0, 1);
      }
    }
    __syncthreads();

    if (j0 <= i0 + 15) {   // wave-uniform causal cutoff (EXEC stays all-ones)
      // --- AC = Qw . K^T : two 16x16 j-tiles
      v8f ac0 = {}, ac1 = {};
      {
        int c = lane & 15;
        Frag bb;
        bb = ldB(Ksm, c,      0, lane); ac0 = wmma_bf16(aw[0], bb, ac0);
        bb = ldB(Ksm, c,      1, lane); ac0 = wmma_bf16(aw[1], bb, ac0);
        bb = ldB(Ksm, 16 + c, 0, lane); ac1 = wmma_bf16(aw[0], bb, ac1);
        bb = ldB(Ksm, 16 + c, 1, lane); ac1 = wmma_bf16(aw[1], bb, ac1);
      }
      // --- BD window: W0 = previous W2; compute W1, W2 for 32 new R rows
      const int u = 497 + j0 - i0;
      v8f w0 = w2acc, w1 = {}, w2n = {};
      {
        int c = lane & 15;
        int s1 = (u + 16 + c) & 255, s2 = (u + 32 + c) & 255;
        Frag bb;
        bb = ldB(ringS, s1, 0, lane); w1  = wmma_bf16(ar[0], bb, w1);
        bb = ldB(ringS, s1, 1, lane); w1  = wmma_bf16(ar[1], bb, w1);
        bb = ldB(ringS, s2, 0, lane); w2n = wmma_bf16(ar[0], bb, w2n);
        bb = ldB(ringS, s2, 1, lane); w2n = wmma_bf16(ar[1], bb, w2n);
      }
      w2acc = w2n;

      // --- spill W (16x48) to per-wave LDS for the rel-shift diagonal gather
      {
        int col = lane & 15, rb = (lane & 16) ? 8 : 0;
        float* wp = Wg[wv];
        #pragma unroll
        for (int v = 0; v < 8; ++v) {
          wp[(v + rb) * 48 + col]      = w0[v];
          wp[(v + rb) * 48 + 16 + col] = w1[v];
          wp[(v + rb) * 48 + 32 + col] = w2n[v];
        }
      }

      // --- assemble S = (AC + BD + EF)*SCALE with causal mask
      float sA[8], sB[8];
      {
        int col = lane & 15, rb = (lane & 16) ? 8 : 0;
        const float* wp = Wg[wv];
        // 32-bit seg_mat addressing; column offsets are loop-invariant
        int sgA = (j0 + col) * BSZ + b;        // + row*(KLEN*BSZ)
        int sgB = sgA + 16 * BSZ;
        const unsigned char* smb = seg_mat + (size_t)i0 * (KLEN * BSZ);
        #pragma unroll
        for (int v = 0; v < 8; ++v) {
          int row = v + rb;
          int gi = i0 + row;
          float e0 = efs[wv * 16 + row][0], e1 = efs[wv * 16 + row][1];
          const float* wq = wp + row * 48 + (15 - row);
          const unsigned char* smr = smb + row * (KLEN * BSZ);
          {
            int gj = j0 + col;
            unsigned char g = smr[sgA];
            float s = (ac0[v] + wq[col] + (g ? e1 : e0)) * SCALEF;
            sA[v] = (gj <= gi) ? s : -3.0e38f;
          }
          {
            int gj = j0 + 16 + col;
            unsigned char g = smr[sgB];
            float s = (ac1[v] + wq[16 + col] + (g ? e1 : e0)) * SCALEF;
            sB[v] = (gj <= gi) ? s : -3.0e38f;
          }
        }
      }

      // --- online softmax (per-row reductions within 16-lane halves)
      #pragma unroll
      for (int v = 0; v < 8; ++v) {
        float x = fmaxf(sA[v], sB[v]);
        #pragma unroll
        for (int d = 1; d < 16; d <<= 1) x = fmaxf(x, __shfl_xor(x, d, 32));
        float mn = fmaxf(mrow[v], x);
        float alpha = __expf(mrow[v] - mn);
        mrow[v] = mn;
        float p0 = __expf(sA[v] - mn);
        float p1 = __expf(sB[v] - mn);
        float ps = p0 + p1;
        #pragma unroll
        for (int d = 1; d < 16; d <<= 1) ps += __shfl_xor(ps, d, 32);
        lrow[v] = lrow[v] * alpha + ps;
        o0[v] *= alpha; o1[v] *= alpha; o2[v] *= alpha; o3[v] *= alpha;
        sA[v] = p0; sB[v] = p1;
      }

      // --- P -> LDS: interleaved K-slots => one b32 store per element pair
      {
        int col = lane & 15, rb = (lane & 16) ? 8 : 0;
        unsigned* pp = Pt[wv];
        #pragma unroll
        for (int v = 0; v < 8; ++v)
          pp[(v + rb) * 20 + col] = pkbf(sA[v], sB[v]);
      }
      Frag pf = ldP(ptu, lane);
      o0 = wmma_bf16(pf, ldV(vtu, 0, lane), o0);
      o1 = wmma_bf16(pf, ldV(vtu, 1, lane), o1);
      o2 = wmma_bf16(pf, ldV(vtu, 2, lane), o2);
      o3 = wmma_bf16(pf, ldV(vtu, 3, lane), o3);
    }
    __syncthreads();
  }

  // ---- epilogue: normalize by row sums and store (coalesced 16-lane rows)
  {
    int col = lane & 15, rb = (lane & 16) ? 8 : 0;
    #pragma unroll
    for (int v = 0; v < 8; ++v) {
      int row = v + rb;
      float inv = 1.0f / lrow[v];
      float* op = out + (size_t)(i0 + row) * ROWSTR + bn;
      op[col]      = o0[v] * inv;
      op[16 + col] = o1[v] * inv;
      op[32 + col] = o2[v] * inv;
      op[48 + col] = o3[v] * inv;
    }
  }
}

extern "C" void kernel_launch(void* const* d_in, const int* in_sizes, int n_in,
                              void* d_out, int out_size, void* d_ws, size_t ws_size,
                              hipStream_t stream) {
  const float*         q   = (const float*)d_in[0];
  const float*         kh  = (const float*)d_in[1];
  const float*         vh  = (const float*)d_in[2];
  const float*         kr  = (const float*)d_in[3];
  const float*         se  = (const float*)d_in[4];
  const unsigned char* sm  = (const unsigned char*)d_in[5];
  const float*         rwb = (const float*)d_in[6];
  const float*         rrb = (const float*)d_in[7];
  const float*         rsb = (const float*)d_in[8];
  // d_in[9] = attn_mask (causal) -- reproduced analytically in-kernel.
  float* out = (float*)d_out;

  dim3 grid(QLEN / BM, BSZ, NHEAD);
  xlnet_rel_attn<<<grid, dim3(256), 0, stream>>>(q, kh, vh, kr, se, sm,
                                                 rwb, rrb, rsb, out);
}
